// FBPINN_7164005450310
// MI455X (gfx1250) — matrix operand order repaired and checked by reading, not compiled, not run on hardware
//
#include <hip/hip_runtime.h>
#include <math.h>

typedef __attribute__((ext_vector_type(8)))  _Float16 v8h;
typedef __attribute__((ext_vector_type(16))) _Float16 v16h;
typedef __attribute__((ext_vector_type(8)))  float    v8f;
typedef __attribute__((ext_vector_type(4)))  int      v4i;

#define K_SUB 16
#define N_PTS 16384
#define W_HID 256
#define TW_F  0.2f
#define PI_F  3.14159265358979f

// Hardware tanh on gfx1250 (v_tanh_f32, TRANS32 class -> co-executes with WMMA).
__device__ __forceinline__ float fast_tanh(float v) {
#if __has_builtin(__builtin_amdgcn_tanhf)
  return __builtin_amdgcn_tanhf(v);
#elif __has_builtin(__builtin_amdgcn_tanh_f32)
  return __builtin_amdgcn_tanh_f32(v);
#else
  return tanhf(v);
#endif
}

// ---- CDNA5 async global->LDS copy (ASYNCcnt-tracked), with safe fallback ----
#if __has_builtin(__builtin_amdgcn_global_load_async_to_lds_b128)
#define HAVE_ASYNC_LDS 1
#endif

typedef __attribute__((address_space(1))) v4i as1_v4i;
typedef __attribute__((address_space(3))) v4i as3_v4i;

__device__ __forceinline__ void copy_b128(const void* g, void* l) {
#ifdef HAVE_ASYNC_LDS
  __builtin_amdgcn_global_load_async_to_lds_b128(
      (as1_v4i*)(void*)g, (as3_v4i*)l, 0, 0);
#else
  *(v8h*)l = *(const v8h*)g;     // synchronous fallback through VGPRs
#endif
}

#ifdef HAVE_ASYNC_LDS
#if __has_builtin(__builtin_amdgcn_s_wait_asynccnt)
#define WAIT_ASYNC(n) __builtin_amdgcn_s_wait_asynccnt(n)
#else
#define WAIT_ASYNC(n) asm volatile("s_wait_asynccnt %0" :: "n"(n) : "memory")
#endif
#else
#define WAIT_ASYNC(n) ((void)0)
#endif

// ---------------- workspace layout (bytes) ----------------
// win  : [N][K]    float   @ 0        (1 MiB)
// W1p  : packed f16 B-frags @ 1 MiB   (2 MiB)
// W2p  : packed f16 B-frags @ 3 MiB   (2 MiB)
// part : [K][N]    float   @ 5 MiB    (1 MiB)

// =============== normalized cosine window weights ===============
__global__ void window_kernel(const float* __restrict__ x,
                              const float* __restrict__ xmins,
                              const float* __restrict__ xmaxs,
                              float* __restrict__ win) {
  int n = blockIdx.x * blockDim.x + threadIdx.x;
  if (n >= N_PTS) return;
  float xd0 = x[2*n+0], xd1 = x[2*n+1];
  float raw[K_SUB];
  float s = 0.f;
#pragma unroll
  for (int k = 0; k < K_SUB; ++k) {
    float p = 1.f;
#pragma unroll
    for (int d = 0; d < 2; ++d) {
      float xv = d ? xd1 : xd0;
      float mn = xmins[2*k+d], mx = xmaxs[2*k+d];
      float tl = (xv - (mn - TW_F)) * (1.f/(2.f*TW_F));
      tl = fminf(fmaxf(tl, 0.f), 1.f);
      float tr = ((mx + TW_F) - xv) * (1.f/(2.f*TW_F));
      tr = fminf(fmaxf(tr, 0.f), 1.f);
      p *= 0.25f * (1.f - __cosf(PI_F*tl)) * (1.f - __cosf(PI_F*tr));
    }
    raw[k] = p;
    s += p;
  }
  float inv = 1.f / (s + 1e-9f);
#pragma unroll
  for (int k = 0; k < K_SUB; ++k) win[n*K_SUB + k] = raw[k]*inv;
}

// =============== repack W1/W2 (f32 [k][w][v]) -> f16 WMMA B fragments ===============
// Fragment f=(k*8+kc)*16+nt holds B tile (K=kc*32..+31, N=nt*16..+15).
// B layout used here (16-bit, 32x16): lane L: N = L&15; half=(L>>4): dword p holds
// rows K = half*16 + 2p (lo16) and +1 (hi16).  Stored lane-contiguous:
// dword index = f*256 + L*8 + p, so a lane reads its v16h as 2x b128.
// A (k,kc) stage = 16 consecutive fragments = 16 KiB contiguous.
__global__ void repack_kernel(const float* __restrict__ W1,
                              const float* __restrict__ W2,
                              unsigned int* __restrict__ W1p,
                              unsigned int* __restrict__ W2p) {
  unsigned int gid = blockIdx.x * blockDim.x + threadIdx.x;   // 0..1048575
  unsigned int layer = gid >> 19;
  unsigned int t  = gid & 0x7FFFFu;
  unsigned int p  = t & 7u;
  unsigned int L  = (t >> 3) & 31u;
  unsigned int nt = (t >> 8) & 15u;
  unsigned int kc = (t >> 12) & 7u;
  unsigned int k  = t >> 15;
  unsigned int w  = kc*32u + (L >> 4)*16u + 2u*p;
  unsigned int v  = nt*16u + (L & 15u);
  const float* src = layer ? W2 : W1;
  unsigned int* dst = layer ? W2p : W1p;
  union { _Float16 h[2]; unsigned int u; } pk;
  pk.h[0] = (_Float16)src[(k*256u + w    )*256u + v];
  pk.h[1] = (_Float16)src[(k*256u + w + 1)*256u + v];
  unsigned int f = (k*8u + kc)*16u + nt;
  dst[f*256u + L*8u + p] = pk.u;
}

// =============== fused per-subdomain MLP ===============
// grid = (N/32, K); block = 64 (2 waves); each wave owns 16 points.
// LDS: H ping-pong 32 KiB + B double-buffer 32 KiB = 64 KiB.
__launch_bounds__(64, 1)
__global__ void fbpinn_main(const float* __restrict__ x,
                            const float* __restrict__ W0,
                            const float* __restrict__ b0,
                            const float* __restrict__ b1,
                            const float* __restrict__ b2,
                            const float* __restrict__ W3,
                            const float* __restrict__ b3,
                            const float* __restrict__ xmins,
                            const float* __restrict__ xmaxs,
                            const _Float16* __restrict__ W1p,
                            const _Float16* __restrict__ W2p,
                            const float* __restrict__ win,
                            float* __restrict__ part) {
  __shared__ __align__(16) _Float16 Hs[2][32][W_HID];   // 32 KiB
  __shared__ __align__(16) _Float16 Bst[2][8192];       // 2 x 16 KiB B stages

  const int tid    = threadIdx.x;        // 0..63
  const int lane   = tid & 31;
  const int wave   = tid >> 5;           // 0..1
  const int k      = blockIdx.y;
  const int nb     = blockIdx.x * 32;
  const int r0     = wave * 16;          // this wave's 16 LDS rows
  const int halfId = lane >> 4;
  const int lane16 = lane & 15;

  // per-subdomain normalization (reciprocal once)
  const float cx0 = 0.5f*(xmins[2*k+0] + xmaxs[2*k+0]);
  const float cx1 = 0.5f*(xmins[2*k+1] + xmaxs[2*k+1]);
  const float is0 = 1.f / fmaxf(0.5f*(xmaxs[2*k+0] - xmins[2*k+0]), 1e-9f);
  const float is1 = 1.f / fmaxf(0.5f*(xmaxs[2*k+1] - xmins[2*k+1]), 1e-9f);

  // ---- layer 0: [16,2] x [2,256] (scalar VALU; tiny) ----
  for (int r = 0; r < 16; ++r) {
    int n = nb + r0 + r;
    float x0 = (x[2*n+0] - cx0) * is0;
    float x1 = (x[2*n+1] - cx1) * is1;
#pragma unroll
    for (int j = 0; j < 8; ++j) {
      int c = lane + 32*j;
      float h = fast_tanh(x0 * W0[(k*2+0)*W_HID + c] +
                          x1 * W0[(k*2+1)*W_HID + c] + b0[k*W_HID + c]);
      Hs[0][r0 + r][c] = (_Float16)h;
    }
  }
  // H is per-wave private (each wave only touches its own 16 rows)

  // ---- layers 1,2: 16x256 += 16x256 * 256x256 via WMMA f16->f32 ----
  for (int layer = 0; layer < 2; ++layer) {
    const _Float16* Wp  = layer ? W2p : W1p;
    const float*    bia = layer ? b2  : b1;
    const int inb = layer, outb = layer ^ 1;

    v8f acc[16];

    // issue async stage for kc=0 into Bst[0]: 16 KiB, 16 x b128 per thread
    {
      const char* src0 = (const char*)Wp + (size_t)(k*8 + 0)*16384 + tid*16;
      char*       dst0 = (char*)&Bst[0][0] + tid*16;
#pragma unroll
      for (int j = 0; j < 16; ++j)
        copy_b128(src0 + j*1024, dst0 + j*1024);
    }

#pragma unroll
    for (int kc = 0; kc < 8; ++kc) {
      // prefetch next stage while computing this one
      if (kc < 7) {
        const char* srcn = (const char*)Wp + (size_t)(k*8 + kc + 1)*16384 + tid*16;
        char*       dstn = (char*)&Bst[(kc + 1) & 1][0] + tid*16;
#pragma unroll
        for (int j = 0; j < 16; ++j)
          copy_b128(srcn + j*1024, dstn + j*1024);
        WAIT_ASYNC(16);     // stage kc done (in-order), stage kc+1 in flight
      } else {
        WAIT_ASYNC(0);
      }
      __syncthreads();      // other wave's stage writes visible

      // A fragment (16x32 f16): lane -> row M=lane16; K = kc*32 + half*16 + s (contiguous, as in B packing).
      const v8h* ap = (const v8h*)&Hs[inb][r0 + lane16][kc*32 + halfId*16];
      v8h alo = ap[0];          // K = kc*32 + half*16 + 0..7
      v8h ahi = ap[1];          // K = kc*32 + half*16 + 8..15
      v16h A = __builtin_shufflevector(alo, ahi,
                 0,1,2,3,4,5,6,7,8,9,10,11,12,13,14,15);

      const _Float16* Bbuf = &Bst[kc & 1][0];
#pragma unroll
      for (int nt = 0; nt < 16; ++nt) {
        const v8h* bp = (const v8h*)(Bbuf + nt*512 + lane*16);
        v8h blo = bp[0];
        v8h bhi = bp[1];
        v16h B = __builtin_shufflevector(blo, bhi,
                   0,1,2,3,4,5,6,7,8,9,10,11,12,13,14,15);
        if (kc == 0) {
          acc[nt] = __builtin_amdgcn_wmma_f32_16x16x32_f16(
              false, A, false, B, (short)0, (v8f){}, false, false);
        } else {
          acc[nt] = __builtin_amdgcn_wmma_f32_16x16x32_f16(
              false, A, false, B, (short)0, acc[nt], false, false);
        }
      }
      __syncthreads();      // both waves done reading Bst[kc&1] before reuse
    }

    // bias + tanh + store back (C layout: lane->N=lane16, VGPR v -> M=v+8*half)
#pragma unroll
    for (int nt = 0; nt < 16; ++nt) {
      float bv = bia[k*W_HID + nt*16 + lane16];
#pragma unroll
      for (int v = 0; v < 8; ++v) {
        float h = fast_tanh(acc[nt][v] + bv);
        Hs[outb][r0 + v + 8*halfId][nt*16 + lane16] = (_Float16)h;
      }
    }
  }

  // ---- layer 3: [16,256] x [256,1] dot + windowed combine ----
  {
    const int m = lane16;
    const v8h*   hp = (const v8h*)&Hs[0][r0 + m][halfId*128];
    const float* w3 = W3 + k*W_HID + halfId*128;
    float s = 0.f;
#pragma unroll
    for (int j = 0; j < 16; ++j) {
      v8h hv = hp[j];
#pragma unroll
      for (int e = 0; e < 8; ++e) s += (float)hv[e] * w3[j*8 + e];
    }
    s += __shfl_xor(s, 16, 32);          // combine the two half-sums
    if (lane < 16) {
      int n = nb + r0 + m;
      float res = s + b3[k];
      part[(size_t)k*N_PTS + n] = win[n*K_SUB + k] * res;
    }
  }
}

// =============== deterministic final reduction over K ===============
__global__ void reduce_kernel(const float* __restrict__ part,
                              float* __restrict__ out) {
  int n = blockIdx.x * blockDim.x + threadIdx.x;
  if (n >= N_PTS) return;
  float s = 0.f;
#pragma unroll
  for (int k = 0; k < K_SUB; ++k) s += part[(size_t)k*N_PTS + n];
  out[n] = s;
}

extern "C" void kernel_launch(void* const* d_in, const int* in_sizes, int n_in,
                              void* d_out, int out_size, void* d_ws, size_t ws_size,
                              hipStream_t stream) {
  const float* x     = (const float*)d_in[0];
  const float* W0    = (const float*)d_in[1];
  const float* b0    = (const float*)d_in[2];
  const float* W1    = (const float*)d_in[3];
  const float* b1    = (const float*)d_in[4];
  const float* W2    = (const float*)d_in[5];
  const float* b2    = (const float*)d_in[6];
  const float* W3    = (const float*)d_in[7];
  const float* b3    = (const float*)d_in[8];
  const float* xmins = (const float*)d_in[9];
  const float* xmaxs = (const float*)d_in[10];
  float* out = (float*)d_out;

  char* ws = (char*)d_ws;
  float*        win  = (float*)(ws);
  unsigned int* W1p  = (unsigned int*)(ws + (1u << 20));
  unsigned int* W2p  = (unsigned int*)(ws + 3u*(1u << 20));
  float*        part = (float*)(ws + 5u*(1u << 20));

  window_kernel<<<N_PTS/256, 256, 0, stream>>>(x, xmins, xmaxs, win);
  repack_kernel<<<4096, 256, 0, stream>>>(W1, W2, W1p, W2p);

  dim3 grid(N_PTS/32, K_SUB);
  fbpinn_main<<<grid, 64, 0, stream>>>(x, W0, b0, b1, b2, W3, b3,
                                       xmins, xmaxs,
                                       (const _Float16*)W1p,
                                       (const _Float16*)W2p,
                                       win, part);

  reduce_kernel<<<N_PTS/256, 256, 0, stream>>>(part, out);
}
